// MultiHeadSelfAttention_20229295964711
// MI455X (gfx1250) — matrix operand order repaired and checked by reading, not compiled, not run on hardware
//
#include <hip/hip_runtime.h>
#include <hip/hip_bf16.h>
#include <stdint.h>

#define D_MODEL   768
#define NUM_HEADS 12
#define HEAD_DIM  64
#define SEQ       4096
#define BATCH     2
#define MROWS     (BATCH * SEQ)   /* 8192 */

typedef __attribute__((ext_vector_type(16))) __bf16   v16bf;
typedef __attribute__((ext_vector_type(8)))  float    v8f;
typedef __attribute__((ext_vector_type(8)))  uint32_t v8u;

static __device__ __forceinline__ uint16_t f2bf(float f) {
  uint32_t u = __builtin_bit_cast(uint32_t, f);
  u += 0x7FFFu + ((u >> 16) & 1u);                 // round-to-nearest-even
  return (uint16_t)(u >> 16);
}
static __device__ __forceinline__ float bf2f(uint16_t h) {
  uint32_t u = ((uint32_t)h) << 16;
  return __builtin_bit_cast(float, u);
}
static __device__ __forceinline__ v8f wmma_bf16(v8u a, v8u b, v8f c) {
  return __builtin_amdgcn_wmma_f32_16x16x32_bf16(
      false, __builtin_bit_cast(v16bf, a),
      false, __builtin_bit_cast(v16bf, b),
      (short)0, c, false, false);
}
// LDS offset = low 32 bits of the generic pointer (flat->LDS aperture truncation)
static __device__ __forceinline__ uint32_t lds_off(const void* p) {
  return (uint32_t)(uintptr_t)p;
}

// ---------------------------------------------------------------- converts
__global__ void f32_to_bf16_kernel(const float* __restrict__ in,
                                   uint16_t* __restrict__ out, int n) {
  int i = blockIdx.x * blockDim.x + threadIdx.x;
  if (i < n) out[i] = f2bf(in[i]);
}

// ---------------------------------------------------------------- GEMM  C = A[M,K] * W[N,K]^T
// wave tile: 32 rows x 64 cols (2 A-frags, 8 accumulators)
// mode 0: write bf16 to [B,H,S,hd] layout (QKV projections)
// mode 1: write fp32 row-major [M,N] (output projection -> d_out)
__global__ void gemm_xwt_kernel(const uint16_t* __restrict__ A,
                                const uint16_t* __restrict__ W,
                                void* __restrict__ out, int mode) {
  const int lane = threadIdx.x & 31;
  const int wave = threadIdx.x >> 5;
  const int hi   = lane >> 4;
  const int l15  = lane & 15;
  const int m0   = blockIdx.x * 256 + wave * 32;
  const int n0   = blockIdx.y * 64;

  v8f acc[2][4] = {};
  const uint16_t* arow0 = A + (size_t)(m0 + l15) * D_MODEL;
  const uint16_t* arow1 = A + (size_t)(m0 + 16 + l15) * D_MODEL;

  for (int k0 = 0; k0 < D_MODEL; k0 += 32) {
    // A fragments: 16x32 bf16, lane(l15)=row, documented K placement per dword
    v8u af[2];
#pragma unroll
    for (int i = 0; i < 8; ++i) {
      int kb = k0 + ((i & 4) ? 16 : 0) + (hi ? 8 : 0) + 2 * (i & 3);
      af[0][i] = *(const uint32_t*)(arow0 + kb);
      af[1][i] = *(const uint32_t*)(arow1 + kb);
    }
#pragma unroll
    for (int t = 0; t < 4; ++t) {
      // B fragment: 32x16, B[k][n] = W[n][k]; lane(l15)=col n, reads row n of W
      const uint16_t* wrow = W + (size_t)(n0 + t * 16 + l15) * D_MODEL;
      v8u bf;
#pragma unroll
      for (int i = 0; i < 8; ++i) {
        int kb = k0 + ((i & 4) ? 16 : 0) + (hi ? 8 : 0) + 2 * (i & 3);
        bf[i] = *(const uint32_t*)(wrow + kb);
      }
      acc[0][t] = wmma_bf16(af[0], bf, acc[0][t]);
      acc[1][t] = wmma_bf16(af[1], bf, acc[1][t]);
    }
  }

  if (mode == 0) {
    uint16_t* o = (uint16_t*)out;
#pragma unroll
    for (int a = 0; a < 2; ++a)
#pragma unroll
      for (int t = 0; t < 4; ++t)
#pragma unroll
        for (int r = 0; r < 8; ++r) {
          int gm = m0 + a * 16 + r + 8 * hi;  // C layout: M = vgpr + 8*hi
          int gn = n0 + t * 16 + l15;         //           N = lane&15
          int b = gm >> 12, s = gm & 4095;    // row -> (batch, seq)
          int h = gn >> 6,  d = gn & 63;      // col -> (head, dim)
          o[(((size_t)(b * NUM_HEADS + h) * SEQ + s) << 6) + d] = f2bf(acc[a][t][r]);
        }
  } else {
    float* o = (float*)out;
#pragma unroll
    for (int a = 0; a < 2; ++a)
#pragma unroll
      for (int t = 0; t < 4; ++t)
#pragma unroll
        for (int r = 0; r < 8; ++r) {
          int gm = m0 + a * 16 + r + 8 * hi;
          int gn = n0 + t * 16 + l15;
          o[(size_t)gm * D_MODEL + gn] = acc[a][t][r];
        }
  }
}

// ---------------------------------------------------------------- RoPE (in-place on bf16 Q,K)
// Q additionally pre-scaled by 1/sqrt(HEAD_DIM) so flash inner loop skips it.
__global__ void rope_inplace_kernel(uint16_t* __restrict__ q,
                                    uint16_t* __restrict__ k,
                                    const int* __restrict__ pos) {
  int tid = blockIdx.x * blockDim.x + threadIdx.x;
  int p  = tid & 31;            // pair index within head_dim (0..31)
  int s  = (tid >> 5) & 4095;   // seq position
  int bh = tid >> 17;           // batch*head
  if (bh >= BATCH * NUM_HEADS) return;
  // inv_freq = 10000^(-2p/64) = exp(-2p/64 * ln(10000))
  float inv = __expf(-(float)(2 * p) * (9.210340371976f / 64.0f));
  float ang = (float)pos[s] * inv;
  float sn, cs;
  __sincosf(ang, &sn, &cs);
  size_t off = ((((size_t)bh << 12) + s) << 6) + 2 * p;
  float q0 = bf2f(q[off]), q1 = bf2f(q[off + 1]);
  q[off]     = f2bf((q0 * cs - q1 * sn) * 0.125f);   // fold softmax scale into Q
  q[off + 1] = f2bf((q1 * cs + q0 * sn) * 0.125f);
  float k0 = bf2f(k[off]), k1 = bf2f(k[off + 1]);
  k[off]     = f2bf(k0 * cs - k1 * sn);
  k[off + 1] = f2bf(k1 * cs + k0 * sn);
}

// ---------------------------------------------------------------- causal flash attention
// grid: (SEQ/128, BATCH*NUM_HEADS), block 256 (8 waves). Wave = 16 query rows.
__global__ void flash_attn_kernel(const uint16_t* __restrict__ q,
                                  const uint16_t* __restrict__ k,
                                  const uint16_t* __restrict__ v,
                                  uint16_t* __restrict__ o) {
  __shared__ uint16_t Ks[32][66];       // 32 keys x 64 dims, padded (bank stagger)
  __shared__ uint16_t Vt[64][34];       // V transposed: dim x key, padded
  __shared__ uint16_t Ps[8][16][34];    // per-wave P staging, padded

  const int lane = threadIdx.x & 31;
  const int wave = threadIdx.x >> 5;
  const int hi   = lane >> 4;
  const int l15  = lane & 15;
  const int bh   = blockIdx.y;
  const int b    = bh / NUM_HEADS, h = bh % NUM_HEADS;
  const int q0   = blockIdx.x * 128;
  const int qw   = q0 + wave * 16;
  const size_t base = (size_t)bh * SEQ * HEAD_DIM;

  // K async-staging addresses for this thread (16B per thread = 4KB chunk)
  const int kkey = threadIdx.x >> 3;            // 0..31
  const int kq8  = (threadIdx.x & 7) * 8;       // column (in shorts), 8-short segs
  const uint32_t ks_loff = lds_off(&Ks[kkey][kq8]);

  // Q tile resident as two A fragments (16 x 64 = 2 x (16x32))
  v8u qa[2];
  {
    const uint16_t* qrow = q + base + (size_t)(qw + l15) * HEAD_DIM;
#pragma unroll
    for (int c = 0; c < 2; ++c)
#pragma unroll
      for (int i = 0; i < 8; ++i) {
        int kb = 32 * c + ((i & 4) ? 16 : 0) + (hi ? 8 : 0) + 2 * (i & 3);
        qa[c][i] = *(const uint32_t*)(qrow + kb);
      }
  }

  v8f oacc[4] = {};
  float mrun[8], lrun[8];
#pragma unroll
  for (int r = 0; r < 8; ++r) { mrun[r] = -1e30f; lrun[r] = 0.0f; }

  const int nchunks = (q0 >> 5) + 4;    // causal: keys up to q0+127
  for (int jc = 0; jc < nchunks; ++jc) {
    const int j = jc * 32;
    __syncthreads();
    // --- K chunk: async global->LDS copy (CDNA5 async path, ASYNCcnt) ---
    {
      const uint16_t* gk = k + base + (size_t)(j + kkey) * 64 + kq8;
      asm volatile("global_load_async_to_lds_b128 %0, %1, off"
                   :: "v"(ks_loff), "v"(gk) : "memory");
    }
    // --- V chunk: transposed staging Vt[dim][key] ---
#pragma unroll
    for (int ii = 0; ii < 4; ++ii) {
      int idx = threadIdx.x * 4 + ii;   // 0..1023 dwords
      int key = idx >> 5;
      int dp  = idx & 31;               // dword within 64-dim row -> dims 2dp,2dp+1
      uint32_t vv = *(const uint32_t*)(v + base + (size_t)(j + key) * 64 + dp * 2);
      Vt[2 * dp][key]     = (uint16_t)vv;
      Vt[2 * dp + 1][key] = (uint16_t)(vv >> 16);
    }
    asm volatile("s_wait_asynccnt 0x0" ::: "memory");
    __syncthreads();
    if (j > qw + 15) continue;          // wave-uniform: chunk fully above diagonal

    // S = Qs*K^T (Q pre-scaled): two 16x16 tiles (keys j..j+15, j+16..j+31)
    v8f st[2];
#pragma unroll
    for (int t = 0; t < 2; ++t) {
      v8f s = {};
#pragma unroll
      for (int c = 0; c < 2; ++c) {
        v8u kb;
#pragma unroll
        for (int i = 0; i < 8; ++i) {
          int d = 32 * c + ((i & 4) ? 16 : 0) + (hi ? 8 : 0) + 2 * (i & 3);
          kb[i] = *(const uint32_t*)&Ks[t * 16 + l15][d];
        }
        s = wmma_bf16(qa[c], kb, s);
      }
      st[t] = s;
    }

    // causal mask only near the diagonal (wave-uniform branch)
    if (j + 31 > qw) {
#pragma unroll
      for (int r = 0; r < 8; ++r) {
        int row = qw + r + 8 * hi;
#pragma unroll
        for (int t = 0; t < 2; ++t) {
          int key = j + t * 16 + l15;
          st[t][r] = (key <= row) ? st[t][r] : -1e30f;
        }
      }
    }

    // chunk row-max (reduce across 16-lane half) + online softmax update
#pragma unroll
    for (int r = 0; r < 8; ++r) {
      float m2 = fmaxf(st[0][r], st[1][r]);
#pragma unroll
      for (int m = 1; m <= 8; m <<= 1)
        m2 = fmaxf(m2, __shfl_xor(m2, m, 32));
      float mnew  = fmaxf(mrun[r], m2);
      float alpha = __expf(mrun[r] - mnew);
      mrun[r] = mnew;
      float p0 = __expf(st[0][r] - mnew);
      float p1 = __expf(st[1][r] - mnew);
      st[0][r] = p0; st[1][r] = p1;
      float rs = p0 + p1;
#pragma unroll
      for (int m = 1; m <= 8; m <<= 1)
        rs += __shfl_xor(rs, m, 32);
      lrun[r] = lrun[r] * alpha + rs;
#pragma unroll
      for (int t4 = 0; t4 < 4; ++t4) oacc[t4][r] *= alpha;
    }

    // stage P (C layout -> LDS) then regather as A fragment (16x32)
#pragma unroll
    for (int t = 0; t < 2; ++t)
#pragma unroll
      for (int r = 0; r < 8; ++r)
        Ps[wave][r + 8 * hi][t * 16 + l15] = f2bf(st[t][r]);
    asm volatile("s_wait_dscnt 0x0" ::: "memory");   // same-wave LDS RAW
    v8u pa;
#pragma unroll
    for (int i = 0; i < 8; ++i) {
      int kb = ((i & 4) ? 16 : 0) + (hi ? 8 : 0) + 2 * (i & 3);
      pa[i] = *(const uint32_t*)&Ps[wave][l15][kb];
    }

    // O += P * V : V B-fragments as dword loads from transposed Vt
#pragma unroll
    for (int t4 = 0; t4 < 4; ++t4) {
      int n = t4 * 16 + l15;            // dim (B-fragment column)
      v8u vb;
#pragma unroll
      for (int i = 0; i < 8; ++i) {
        int kk = ((i & 4) ? 16 : 0) + (hi ? 8 : 0) + 2 * (i & 3); // packed key pair
        vb[i] = *(const uint32_t*)&Vt[n][kk];
      }
      oacc[t4] = wmma_bf16(pa, vb, oacc[t4]);
    }
  }

  // normalize and store O as bf16 in [B, S, H*hd] (GEMM-ready)
#pragma unroll
  for (int t4 = 0; t4 < 4; ++t4)
#pragma unroll
    for (int r = 0; r < 8; ++r) {
      int row = qw + r + 8 * hi;
      int d   = t4 * 16 + l15;
      float val = oacc[t4][r] / lrun[r];
      o[((size_t)b * SEQ + row) * D_MODEL + h * HEAD_DIM + d] = f2bf(val);
    }
}

// ---------------------------------------------------------------- launch
extern "C" void kernel_launch(void* const* d_in, const int* in_sizes, int n_in,
                              void* d_out, int out_size, void* d_ws, size_t ws_size,
                              hipStream_t stream) {
  (void)in_sizes; (void)n_in; (void)out_size; (void)ws_size;
  const float* x   = (const float*)d_in[0];
  const int*   pos = (const int*)  d_in[1];
  const float* wq  = (const float*)d_in[2];
  const float* wk  = (const float*)d_in[3];
  const float* wv  = (const float*)d_in[4];
  const float* wo  = (const float*)d_in[5];

  // workspace carve-up (all bf16), every region fully written each call
  uint16_t* p = (uint16_t*)d_ws;
  const size_t NX = (size_t)MROWS * D_MODEL;
  const size_t NW = (size_t)D_MODEL * D_MODEL;
  const size_t NH = (size_t)BATCH * NUM_HEADS * SEQ * HEAD_DIM; // == NX
  uint16_t* x_bf  = p; p += NX;
  uint16_t* wq_bf = p; p += NW;
  uint16_t* wk_bf = p; p += NW;
  uint16_t* wv_bf = p; p += NW;
  uint16_t* wo_bf = p; p += NW;
  uint16_t* q_bf  = p; p += NH;
  uint16_t* k_bf  = p; p += NH;
  uint16_t* v_bf  = p; p += NH;
  uint16_t* o_bf  = p;

  f32_to_bf16_kernel<<<(int)((NX + 255) / 256), 256, 0, stream>>>(x,  x_bf,  (int)NX);
  f32_to_bf16_kernel<<<(int)((NW + 255) / 256), 256, 0, stream>>>(wq, wq_bf, (int)NW);
  f32_to_bf16_kernel<<<(int)((NW + 255) / 256), 256, 0, stream>>>(wk, wk_bf, (int)NW);
  f32_to_bf16_kernel<<<(int)((NW + 255) / 256), 256, 0, stream>>>(wv, wv_bf, (int)NW);
  f32_to_bf16_kernel<<<(int)((NW + 255) / 256), 256, 0, stream>>>(wo, wo_bf, (int)NW);

  dim3 gg(MROWS / 256, D_MODEL / 64);
  gemm_xwt_kernel<<<gg, 256, 0, stream>>>(x_bf, wq_bf, (void*)q_bf, 0);
  gemm_xwt_kernel<<<gg, 256, 0, stream>>>(x_bf, wk_bf, (void*)k_bf, 0);
  gemm_xwt_kernel<<<gg, 256, 0, stream>>>(x_bf, wv_bf, (void*)v_bf, 0);

  int nr = BATCH * NUM_HEADS * SEQ * 32;
  rope_inplace_kernel<<<nr / 256, 256, 0, stream>>>(q_bf, k_bf, pos);

  dim3 gf(SEQ / 128, BATCH * NUM_HEADS);
  flash_attn_kernel<<<gf, 256, 0, stream>>>(q_bf, k_bf, v_bf, o_bf);

  gemm_xwt_kernel<<<gg, 256, 0, stream>>>(o_bf, wo_bf, d_out, 1);
}
